// Model_30081950941534
// MI455X (gfx1250) — 8-bit matrix operand order repaired and checked by reading, compiled, not run on hardware
//
#include <hip/hip_runtime.h>
#include <hip/hip_bf16.h>

typedef __attribute__((ext_vector_type(8))) int v8i;

#define QMAXF 127.0f

__global__ void quantize_rows(const float* __restrict__ in,
                              signed char* __restrict__ q,
                              float* __restrict__ scale,
                              int K) {
    const int row = blockIdx.x;
    const int tid = threadIdx.x;
    const float* rp = in + (size_t)row * K;
    const int base = tid * 16;

    float4 v[4];
    float m = 0.0f;
#pragma unroll
    for (int i = 0; i < 4; ++i) {
        v[i] = *(const float4*)(rp + base + i * 4);
        m = fmaxf(m, fmaxf(fmaxf(fabsf(v[i].x), fabsf(v[i].y)),
                           fmaxf(fabsf(v[i].z), fabsf(v[i].w))));
    }

    __shared__ float red[256];
    red[tid] = m;
    __syncthreads();
#pragma unroll
    for (int s = 128; s > 0; s >>= 1) {
        if (tid < s) red[tid] = fmaxf(red[tid], red[tid + s]);
        __syncthreads();
    }
    const float absmax = fmaxf(red[0], 1e-8f);
    const float sc  = absmax / QMAXF;
    const float inv = QMAXF / absmax;
    if (tid == 0) scale[row] = sc;

    int packed[4];
#pragma unroll
    for (int i = 0; i < 4; ++i) {
        const float* f = (const float*)&v[i];
        int b[4];
#pragma unroll
        for (int j = 0; j < 4; ++j) {
            int qi = (int)rintf(f[j] * inv);
            qi = min(127, max(-127, qi));
            b[j] = qi & 0xFF;
        }
        packed[i] = b[0] | (b[1] << 8) | (b[2] << 16) | (b[3] << 24);
    }
    *(int4*)(q + (size_t)row * K + base) = *(int4*)packed;
}

__global__ void __launch_bounds__(256)
gemm_i8(const signed char* __restrict__ qx,
        const signed char* __restrict__ qw,
        const float* __restrict__ sx,
        const float* __restrict__ sw,
        const float* __restrict__ bias,
        float* __restrict__ out,
        int K, int dout) {
    __shared__ __attribute__((aligned(16))) signed char A_lds[256][64];
    __shared__ __attribute__((aligned(16))) signed char B_lds[128][64];

    const int tid  = threadIdx.x;
    const int lane = tid & 31;
    const int wid  = tid >> 5;
    const int wm   = wid >> 1;
    const int wn   = wid & 1;

    const int t0 = blockIdx.y * 256;
    const int o0 = blockIdx.x * 128;

    v8i C[4][4];
#pragma unroll
    for (int mi = 0; mi < 4; ++mi)
#pragma unroll
        for (int ni = 0; ni < 4; ++ni)
#pragma unroll
            for (int j = 0; j < 8; ++j) C[mi][ni][j] = 0;

    const int a_kb = (lane < 16) ? 0 : 16;
    const int b_kb = (lane < 16) ? 0 : 16;
    const int ln16 = lane & 15;

    for (int k0 = 0; k0 < K; k0 += 64) {
#pragma unroll
        for (int rep = 0; rep < 4; ++rep) {
            const int e   = tid + rep * 256;
            const int row = e >> 2;
            const int off = (e & 3) << 4;
            *(int4*)&A_lds[row][off] =
                *(const int4*)(qx + (size_t)(t0 + row) * K + k0 + off);
        }
#pragma unroll
        for (int rep = 0; rep < 2; ++rep) {
            const int e   = tid + rep * 256;
            const int row = e >> 2;
            const int off = (e & 3) << 4;
            *(int4*)&B_lds[row][off] =
                *(const int4*)(qw + (size_t)(o0 + row) * K + k0 + off);
        }
        __syncthreads();

        v8i B[4];
#pragma unroll
        for (int ni = 0; ni < 4; ++ni) {
            const int orow = 64 * wn + 16 * ni + ln16;
            const signed char* bp = &B_lds[orow][b_kb];
            int4 lo = *(const int4*)(bp);
            int4 hi = *(const int4*)(bp + 32);
            B[ni][0] = lo.x; B[ni][1] = lo.y; B[ni][2] = lo.z; B[ni][3] = lo.w;
            B[ni][4] = hi.x; B[ni][5] = hi.y; B[ni][6] = hi.z; B[ni][7] = hi.w;
        }

#pragma unroll
        for (int mi = 0; mi < 4; ++mi) {
            const int arow = 64 * wm + 16 * mi + ln16;
            const signed char* ap = &A_lds[arow][a_kb];
            v8i A;
#pragma unroll
            for (int g = 0; g < 4; ++g) {
                int2 t = *(const int2*)(ap + 8 * g + 16 * (g >> 1));
                A[2 * g]     = t.x;
                A[2 * g + 1] = t.y;
            }
#pragma unroll
            for (int ni = 0; ni < 4; ++ni) {
                C[mi][ni] = __builtin_amdgcn_wmma_i32_16x16x64_iu8(
                    true, A, true, B[ni], C[mi][ni], false, false);
            }
        }
        __syncthreads();
    }

    const int half = lane >> 4;
#pragma unroll
    for (int ni = 0; ni < 4; ++ni) {
        const int col = o0 + 64 * wn + 16 * ni + ln16;
        const float swv = sw[col];
        const float bv  = bias[col];
#pragma unroll
        for (int mi = 0; mi < 4; ++mi) {
            const int rbase = t0 + 64 * wm + 16 * mi + 8 * half;
#pragma unroll
            for (int j = 0; j < 8; ++j) {
                const int row = rbase + j;
                out[(size_t)row * dout + col] =
                    (float)C[mi][ni][j] * (sx[row] * swv) + bv;
            }
        }
    }
}

extern "C" void kernel_launch(void* const* d_in, const int* in_sizes, int n_in,
                              void* d_out, int out_size, void* d_ws, size_t ws_size,
                              hipStream_t stream) {
    const float* x    = (const float*)d_in[0];
    const float* W    = (const float*)d_in[1];
    const float* bias = (const float*)d_in[2];

    const int dout   = in_sizes[2];
    const int din    = in_sizes[1] / dout;
    const int tokens = in_sizes[0] / din;
    float* out = (float*)d_out;

    signed char* qx = (signed char*)d_ws;
    signed char* qw = qx + (size_t)tokens * din;
    float* sx = (float*)(qw + (size_t)dout * din);
    float* sw = sx + tokens;

    quantize_rows<<<tokens, 256, 0, stream>>>(x, qx, sx, din);
    quantize_rows<<<dout,   256, 0, stream>>>(W, qw, sw, din);

    dim3 grid(dout / 128, tokens / 256);
    gemm_i8<<<grid, 256, 0, stream>>>(qx, qw, sx, sw, bias, out, din, dout);
}
